// EnokeeEncoder_66048007078354
// MI455X (gfx1250) — matrix operand order repaired and checked by reading, not compiled, not run on hardware
//
#include <hip/hip_runtime.h>
#include <hip/hip_bf16.h>

typedef __attribute__((ext_vector_type(16))) _Float16 v16h;
typedef __attribute__((ext_vector_type(8)))  _Float16 v8h;
typedef __attribute__((ext_vector_type(8)))  float    v8f;

#define B_  16
#define M_  128
#define L_  32
#define S_  512
#define D_  768
#define H_  12
#define DFF_ 3072
#define NL_ 4
#define NE_ 50000
#define HD_ 64
#define T_  (B_ * M_)       // 2048 tokens
#define BH_ (B_ * H_)       // 192 attention batches
#define KPAD_ 128           // padded K for classifier stage 2

// ---------------------------------------------------------------------------
// CDNA5 async global->LDS copy (16B per lane), tracked by ASYNCcnt.
// ---------------------------------------------------------------------------
__device__ __forceinline__ void async_copy_b128(unsigned lds_off, const void* gptr)
{
    asm volatile("global_load_async_to_lds_b128 %0, %1, off"
                 :: "v"(lds_off), "v"(gptr) : "memory");
}
__device__ __forceinline__ void wait_async0()
{
    asm volatile("s_wait_asynccnt 0x0" ::: "memory");
}
__device__ __forceinline__ void wait_async4()
{
    asm volatile("s_wait_asynccnt 0x4" ::: "memory");
}

// ---------------------------------------------------------------------------
// f32 -> f16 conversion kernels (weights, once per launch)
// ---------------------------------------------------------------------------
__global__ __launch_bounds__(256)
void cvt_f16(const float* __restrict__ src, _Float16* __restrict__ dst, long n)
{
    long i = (long)blockIdx.x * 256 + threadIdx.x;
    if (i < n) dst[i] = (_Float16)src[i];
}

__global__ __launch_bounds__(256)
void cvt_f16_pad(const float* __restrict__ src, _Float16* __restrict__ dst,
                 int rows, int K, int Kpad)
{
    long i = (long)blockIdx.x * 256 + threadIdx.x;
    long n = (long)rows * Kpad;
    if (i >= n) return;
    int k = (int)(i % Kpad);
    long r = i / Kpad;
    dst[i] = (k < K) ? (_Float16)src[r * K + k] : (_Float16)0.f;
}

// ---------------------------------------------------------------------------
// Mention pooling -> x (f32) and xh (f16). One block (256 threads) per (b,m).
// ---------------------------------------------------------------------------
__global__ __launch_bounds__(256)
void pool_kernel(const float* __restrict__ lhs, const int* __restrict__ pos,
                 const int* __restrict__ amask, const float* __restrict__ attn_w,
                 const float* __restrict__ attn_b,
                 float* __restrict__ x, _Float16* __restrict__ xh)
{
    int bm = blockIdx.x;
    int b  = bm >> 7;
    __shared__ float red[256];
    __shared__ float logits[L_];
    __shared__ float wl[L_];
    __shared__ int   s_cnt;
    int t = threadIdx.x;

    if (t == 0) {
        int cnt = 0;
        const int* p = pos + (size_t)bm * L_;
        while (cnt < L_ && p[cnt] != -1) cnt++;
        if (amask[bm] == 0) cnt = 0;
        s_cnt = cnt;
    }
    __syncthreads();
    int cnt = s_cnt;

    int l = t >> 3, s = t & 7;
    float part = 0.f;
    if (l < cnt) {
        const float* row = lhs + (size_t)b * S_ * D_ + (size_t)l * D_;
        for (int d = s; d < D_; d += 8) part += row[d] * attn_w[d];
    }
    red[t] = part;
    __syncthreads();
    if (s == 0) {
        float v = 0.f;
        #pragma unroll
        for (int i = 0; i < 8; i++) v += red[l * 8 + i];
        logits[l] = ((l < cnt) ? v : 0.f) + attn_b[0];
    }
    __syncthreads();
    if (t < 32) {
        float v  = logits[t];
        float mx = v;
        #pragma unroll
        for (int o = 16; o > 0; o >>= 1) mx = fmaxf(mx, __shfl_xor(mx, o, 32));
        float e  = __expf(v - mx);
        float sm = e;
        #pragma unroll
        for (int o = 16; o > 0; o >>= 1) sm += __shfl_xor(sm, o, 32);
        wl[t] = (t < cnt) ? (e / sm) : 0.f;
    }
    __syncthreads();
    for (int d = t; d < D_; d += 256) {
        float acc = 0.f;
        #pragma unroll 4
        for (int ll = 0; ll < L_; ll++)
            acc += wl[ll] * lhs[(size_t)b * S_ * D_ + (size_t)ll * D_ + d];
        x[(size_t)bm * D_ + d]  = acc;
        xh[(size_t)bm * D_ + d] = (_Float16)acc;
    }
}

// ---------------------------------------------------------------------------
// Batched WMMA GEMM, all-f16 operands: C = act(A @ W^T [+ bias])
//   A: f16 [Nrows,K] (lda mult of 8, K mult of 32)
//   W: f16 [Ncols,K] (ldw mult of 8)
//   OUT: 0 -> f32 stores, 1 -> f16 stores (COLCHK pads OOB cols with 0)
//   C batch base = (bz/cDiv)*sCo + (bz%cDiv)*sCi
// 128 threads = 4 waves (2x2), 64x64 block tile, 32x32 per wave.
// Double-buffered global_load_async_to_lds_b128 staging, last K-step peeled
// so the steady-state loop body is branch-free.
// ---------------------------------------------------------------------------
template <int EPI, bool COLCHK, bool BIAS, int OUT>
__global__ __launch_bounds__(128)
void gemm_h(const _Float16* __restrict__ A, const _Float16* __restrict__ W,
            const float* __restrict__ bias, void* __restrict__ Cout,
            int Ncols, int K,
            long sA, long sW, int cDiv, long sCo, long sCi,
            int lda, int ldw, int ldc)
{
    __shared__ _Float16 As[2][64][40];   // [buf][row][k], 80B rows (16B aligned)
    __shared__ _Float16 Ws[2][64][40];   // [buf][col][k]

    int bz = blockIdx.z;
    A += (size_t)bz * sA;
    W += (size_t)bz * sW;
    size_t cOff = (size_t)(bz / cDiv) * sCo + (size_t)(bz % cDiv) * sCi;

    int rowBase = blockIdx.y * 64;
    int colBase = blockIdx.x * 64;
    int t    = threadIdx.x;
    int lane = t & 31;
    int wid  = t >> 5;
    int wr   = (wid >> 1) * 32;
    int wc   = (wid & 1) * 32;
    int half = lane >> 4, l16 = lane & 15;

    // per-thread staging chunk coordinates (16B = 8 halves per chunk)
    int r0 = t >> 2,           c0 = (t & 3) * 8;
    int r1 = (t + 128) >> 2,   c1 = ((t + 128) & 3) * 8;

    const _Float16* Ar0 = A + (size_t)(rowBase + r0) * lda + c0;
    const _Float16* Ar1 = A + (size_t)(rowBase + r1) * lda + c1;
    const _Float16* Wr0 = W + (size_t)(colBase + r0) * ldw + c0;
    const _Float16* Wr1 = W + (size_t)(colBase + r1) * ldw + c1;
    bool w0ok = (!COLCHK) || (colBase + r0) < Ncols;
    bool w1ok = (!COLCHK) || (colBase + r1) < Ncols;

    auto stage = [&](int buf, int k0) {
        async_copy_b128((unsigned)(size_t)&As[buf][r0][c0], Ar0 + k0);
        async_copy_b128((unsigned)(size_t)&As[buf][r1][c1], Ar1 + k0);
        if (w0ok) async_copy_b128((unsigned)(size_t)&Ws[buf][r0][c0], Wr0 + k0);
        else      { v8h z = {}; *(v8h*)&Ws[buf][r0][c0] = z; }
        if (w1ok) async_copy_b128((unsigned)(size_t)&Ws[buf][r1][c1], Wr1 + k0);
        else      { v8h z = {}; *(v8h*)&Ws[buf][r1][c1] = z; }
    };

    v8f acc[2][2] = {};

    auto compute = [&](int buf) {
        // fragments per CDNA5 16-bit A/B layouts
        v16h af[2], bf[2];
        #pragma unroll
        for (int i = 0; i < 2; i++) {
            int row = wr + i * 16 + l16;
            v8h lo = *(const v8h*)&As[buf][row][half * 8];
            v8h hi = *(const v8h*)&As[buf][row][16 + half * 8];
            #pragma unroll
            for (int j = 0; j < 8; j++) { af[i][j] = lo[j]; af[i][j + 8] = hi[j]; }
            int col = wc + i * 16 + l16;
            v8h blo = *(const v8h*)&Ws[buf][col][half * 8];
            v8h bhi = *(const v8h*)&Ws[buf][col][16 + half * 8];
            #pragma unroll
            for (int j = 0; j < 8; j++) { bf[i][j] = blo[j]; bf[i][j + 8] = bhi[j]; }
        }
        #pragma unroll
        for (int i = 0; i < 2; i++)
            #pragma unroll
            for (int j = 0; j < 2; j++)
                acc[i][j] = __builtin_amdgcn_wmma_f32_16x16x32_f16(
                    false, af[i], false, bf[j], (short)0, acc[i][j], false, false);
    };

    int nk = K >> 5;
    stage(0, 0);

    int kt = 0;
    for (; kt < nk - 1; kt++) {          // steady state: branch-free body
        stage((kt + 1) & 1, (kt + 1) << 5);
        if (COLCHK) wait_async0();       // wave may issue <4/tile -> conservative
        else        wait_async4();       // exactly 4/tile in flight ahead
        __syncthreads();
        compute(kt & 1);
        __syncthreads();
    }
    // peeled last K-step
    wait_async0();
    __syncthreads();
    compute(kt & 1);

    // ---- epilogue: C/D layout row = e + 8*half, col = l16 ----
    float*    Cf = (float*)Cout;
    _Float16* Ch = (_Float16*)Cout;
    #pragma unroll
    for (int i = 0; i < 2; i++) {
        #pragma unroll
        for (int j = 0; j < 2; j++) {
            int col = colBase + wc + j * 16 + l16;
            bool inc = (!COLCHK) || (col < Ncols);
            float bv = (BIAS && inc) ? bias[col] : 0.f;
            #pragma unroll
            for (int e = 0; e < 8; e++) {
                int row = rowBase + wr + i * 16 + half * 8 + e;
                float v = acc[i][j][e] + bv;
                if (EPI == 1) v = fmaxf(v, 0.f);
                if (OUT == 0) {
                    if (inc) Cf[cOff + (size_t)row * ldc + col] = v;
                } else {
                    Ch[cOff + (size_t)row * ldc + col] = (_Float16)(inc ? v : 0.f);
                }
            }
        }
    }
}

// ---------------------------------------------------------------------------
// QKV split (f16): q,k -> [B,H,M,HD]; v -> transposed [B,H,HD,M]
// ---------------------------------------------------------------------------
__global__ __launch_bounds__(256)
void qkv_split(const _Float16* __restrict__ qkv, _Float16* __restrict__ q,
               _Float16* __restrict__ k, _Float16* __restrict__ vT)
{
    size_t idx = (size_t)blockIdx.x * 256 + threadIdx.x;
    if (idx >= (size_t)T_ * 3 * D_) return;
    int    c  = (int)(idx % (3 * D_));
    size_t tm = idx / (3 * D_);
    int b = (int)(tm >> 7), m = (int)(tm & 127);
    _Float16 v = qkv[idx];
    int which = c / D_, cd = c % D_;
    int h = cd / HD_, d = cd % HD_;
    size_t bh = (size_t)b * H_ + h;
    if (which == 0)      q[(bh * M_ + m) * HD_ + d] = v;
    else if (which == 1) k[(bh * M_ + m) * HD_ + d] = v;
    else                 vT[(bh * HD_ + d) * M_ + m] = v;
}

// ---------------------------------------------------------------------------
// Scaled softmax: f32 in -> f16 probabilities out. One block per row of 128.
// ---------------------------------------------------------------------------
__global__ __launch_bounds__(128)
void softmax_scale(const float* __restrict__ S, _Float16* __restrict__ P, float scale)
{
    __shared__ float red[128];
    const float* row = S + (size_t)blockIdx.x * 128;
    int t = threadIdx.x;
    float v = row[t] * scale;
    red[t] = v; __syncthreads();
    for (int o = 64; o > 0; o >>= 1) { if (t < o) red[t] = fmaxf(red[t], red[t + o]); __syncthreads(); }
    float mx = red[0]; __syncthreads();
    float e = __expf(v - mx);
    red[t] = e; __syncthreads();
    for (int o = 64; o > 0; o >>= 1) { if (t < o) red[t] += red[t + o]; __syncthreads(); }
    P[(size_t)blockIdx.x * 128 + t] = (_Float16)(e / red[0]);
}

// ---------------------------------------------------------------------------
// out = LayerNorm(x + delta) * w + b  -> f32 and f16 copies.
// ---------------------------------------------------------------------------
__global__ __launch_bounds__(256)
void add_ln(const float* __restrict__ x, const float* __restrict__ delta,
            const float* __restrict__ w, const float* __restrict__ b,
            float* __restrict__ outF, _Float16* __restrict__ outH)
{
    __shared__ float sv[D_];
    __shared__ float red[256];
    int row = blockIdx.x, t = threadIdx.x;
    const float* xr = x + (size_t)row * D_;
    const float* dr = delta + (size_t)row * D_;
    float sum = 0.f, sq = 0.f;
    for (int d = t; d < D_; d += 256) {
        float v = xr[d] + dr[d];
        sv[d] = v; sum += v; sq += v * v;
    }
    red[t] = sum; __syncthreads();
    for (int o = 128; o > 0; o >>= 1) { if (t < o) red[t] += red[t + o]; __syncthreads(); }
    float mu = red[0] / D_; __syncthreads();
    red[t] = sq; __syncthreads();
    for (int o = 128; o > 0; o >>= 1) { if (t < o) red[t] += red[t + o]; __syncthreads(); }
    float var  = red[0] / D_ - mu * mu;
    float rstd = rsqrtf(var + 1e-5f);
    for (int d = t; d < D_; d += 256) {
        float v = (sv[d] - mu) * rstd * w[d] + b[d];
        outF[(size_t)row * D_ + d] = v;
        outH[(size_t)row * D_ + d] = (_Float16)v;
    }
}

// ---------------------------------------------------------------------------
extern "C" void kernel_launch(void* const* d_in, const int* in_sizes, int n_in,
                              void* d_out, int out_size, void* d_ws, size_t ws_size,
                              hipStream_t stream)
{
    const float* lhs    = (const float*)d_in[0];
    const int*   pos    = (const int*)d_in[1];
    const int*   amask  = (const int*)d_in[2];
    const float* attn_w = (const float*)d_in[3];
    const float* attn_b = (const float*)d_in[4];
    const float* qkv_w  = (const float*)d_in[5];
    const float* qkv_b  = (const float*)d_in[6];
    const float* out_w  = (const float*)d_in[7];
    const float* out_b  = (const float*)d_in[8];
    const float* ln1_w  = (const float*)d_in[9];
    const float* ln1_b  = (const float*)d_in[10];
    const float* ff1_w  = (const float*)d_in[11];
    const float* ff1_b  = (const float*)d_in[12];
    const float* ff2_w  = (const float*)d_in[13];
    const float* ff2_b  = (const float*)d_in[14];
    const float* ln2_w  = (const float*)d_in[15];
    const float* ln2_b  = (const float*)d_in[16];
    const float* cls_w1 = (const float*)d_in[17];
    const float* cls_w2 = (const float*)d_in[18];
    const float* cls_b2 = (const float*)d_in[19];

    // ---- workspace carving (256B aligned) ----
    char* base = (char*)d_ws;
    auto alloc = [&](size_t bytes) -> char* {
        char* r = base; base += (bytes + 255) & ~(size_t)255; return r;
    };
    float* x    = (float*)alloc((size_t)T_ * D_ * 4);
    float* xa   = (float*)alloc((size_t)T_ * D_ * 4);
    float* proj = (float*)alloc((size_t)T_ * D_ * 4);       // also ff2 out
    float* sc   = (float*)alloc((size_t)BH_ * M_ * M_ * 4);

    _Float16* xh    = (_Float16*)alloc((size_t)T_ * D_ * 2);
    _Float16* xah   = (_Float16*)alloc((size_t)T_ * D_ * 2);
    _Float16* qkvbh = (_Float16*)alloc((size_t)T_ * 3 * D_ * 2);
    _Float16* qh    = (_Float16*)alloc((size_t)T_ * D_ * 2);
    _Float16* kh    = (_Float16*)alloc((size_t)T_ * D_ * 2);
    _Float16* vTh   = (_Float16*)alloc((size_t)T_ * D_ * 2);
    _Float16* sch   = (_Float16*)alloc((size_t)BH_ * M_ * M_ * 2);
    _Float16* aoh   = (_Float16*)alloc((size_t)T_ * D_ * 2);
    _Float16* ffh   = (_Float16*)alloc((size_t)T_ * DFF_ * 2);
    _Float16* h100h = (_Float16*)alloc((size_t)T_ * KPAD_ * 2);

    _Float16* wqkvh = (_Float16*)alloc((size_t)NL_ * 3 * D_ * D_ * 2);
    _Float16* wouth = (_Float16*)alloc((size_t)NL_ * D_ * D_ * 2);
    _Float16* wff1h = (_Float16*)alloc((size_t)NL_ * DFF_ * D_ * 2);
    _Float16* wff2h = (_Float16*)alloc((size_t)NL_ * D_ * DFF_ * 2);
    _Float16* wc1h  = (_Float16*)alloc((size_t)100 * D_ * 2);
    _Float16* wc2h  = (_Float16*)alloc((size_t)NE_ * KPAD_ * 2);

    // ---- weight conversion (f32 -> f16, cls_w2 zero-padded K 100->128) ----
    {
        long n;
        n = (long)NL_ * 3 * D_ * D_;
        cvt_f16<<<(n + 255) / 256, 256, 0, stream>>>(qkv_w, wqkvh, n);
        n = (long)NL_ * D_ * D_;
        cvt_f16<<<(n + 255) / 256, 256, 0, stream>>>(out_w, wouth, n);
        n = (long)NL_ * DFF_ * D_;
        cvt_f16<<<(n + 255) / 256, 256, 0, stream>>>(ff1_w, wff1h, n);
        n = (long)NL_ * D_ * DFF_;
        cvt_f16<<<(n + 255) / 256, 256, 0, stream>>>(ff2_w, wff2h, n);
        n = (long)100 * D_;
        cvt_f16<<<(n + 255) / 256, 256, 0, stream>>>(cls_w1, wc1h, n);
        n = (long)NE_ * KPAD_;
        cvt_f16_pad<<<(n + 255) / 256, 256, 0, stream>>>(cls_w2, wc2h, NE_, 100, KPAD_);
    }

    // ---- mention pooling ----
    pool_kernel<<<T_, 256, 0, stream>>>(lhs, pos, amask, attn_w, attn_b, x, xh);

    // ---- transformer layers ----
    for (int i = 0; i < NL_; i++) {
        // qkv: [2048, 2304] f16, +bias
        gemm_h<0, false, true, 1><<<dim3(3 * D_ / 64, T_ / 64, 1), 128, 0, stream>>>(
            xh, wqkvh + (size_t)i * 3 * D_ * D_, qkv_b + (size_t)i * 3 * D_,
            qkvbh, 3 * D_, D_, 0, 0, 1, 0, 0, D_, D_, 3 * D_);

        qkv_split<<<((size_t)T_ * 3 * D_ + 255) / 256, 256, 0, stream>>>(qkvbh, qh, kh, vTh);

        // scores = q @ k^T : [128,128] f32, K=64, 192 batches
        gemm_h<0, false, false, 0><<<dim3(2, 2, BH_), 128, 0, stream>>>(
            qh, kh, nullptr, sc,
            M_, HD_, (long)M_ * HD_, (long)M_ * HD_, 1, (long)M_ * M_, 0,
            HD_, HD_, M_);

        softmax_scale<<<BH_ * M_, 128, 0, stream>>>(sc, sch, 0.125f);

        // ao[b,m, h*HD+d] = P @ V (heads merged via batch split), f16 out
        gemm_h<0, false, false, 1><<<dim3(1, 2, BH_), 128, 0, stream>>>(
            sch, vTh, nullptr, aoh,
            HD_, M_, (long)M_ * M_, (long)HD_ * M_, H_, (long)M_ * D_, (long)HD_,
            M_, M_, D_);

        // output projection -> proj f32, +bias
        gemm_h<0, false, true, 0><<<dim3(D_ / 64, T_ / 64, 1), 128, 0, stream>>>(
            aoh, wouth + (size_t)i * D_ * D_, out_b + (size_t)i * D_,
            proj, D_, D_, 0, 0, 1, 0, 0, D_, D_, D_);

        add_ln<<<T_, 256, 0, stream>>>(x, proj, ln1_w + (size_t)i * D_, ln1_b + (size_t)i * D_, xa, xah);

        // FF1 + relu -> ffh f16, +bias
        gemm_h<1, false, true, 1><<<dim3(DFF_ / 64, T_ / 64, 1), 128, 0, stream>>>(
            xah, wff1h + (size_t)i * DFF_ * D_, ff1_b + (size_t)i * DFF_,
            ffh, DFF_, D_, 0, 0, 1, 0, 0, D_, D_, DFF_);

        // FF2 -> proj f32 (reuse), +bias
        gemm_h<0, false, true, 0><<<dim3(D_ / 64, T_ / 64, 1), 128, 0, stream>>>(
            ffh, wff2h + (size_t)i * D_ * DFF_, ff2_b + (size_t)i * D_,
            proj, D_, DFF_, 0, 0, 1, 0, 0, DFF_, DFF_, D_);

        add_ln<<<T_, 256, 0, stream>>>(xa, proj, ln2_w + (size_t)i * D_, ln2_b + (size_t)i * D_, x, xh);
    }

    // ---- classifier ----
    // h100h: [2048, 128] f16, cols 100..127 zero-filled by epilogue
    gemm_h<0, true, false, 1><<<dim3(2, T_ / 64, 1), 128, 0, stream>>>(
        xh, wc1h, nullptr, h100h,
        100, D_, 0, 0, 1, 0, 0, D_, D_, KPAD_);

    gemm_h<0, true, true, 0><<<dim3((NE_ + 63) / 64, T_ / 64, 1), 128, 0, stream>>>(
        h100h, wc2h, cls_b2, (float*)d_out,
        NE_, KPAD_, 0, 0, 1, 0, 0, KPAD_, KPAD_, NE_);
}
